// WeightOnlyInt4Linear_2130303779339
// MI455X (gfx1250) — matrix operand order repaired and checked by reading, not compiled, not run on hardware
//
#include <hip/hip_runtime.h>
#include <hip/hip_bf16.h>

// ---------------------------------------------------------------------------
// WeightOnlyInt4Linear on gfx1250:
//   Pass 1a: x f32 -> bf16            (ws)
//   Pass 1b: wq int4-code -> bf16     (ws)   w = (q-8)*scale + zero
//   Pass 2 : bf16 GEMM, v_wmma_f32_16x16x32_bf16, async global->LDS staging
//            (global_load_async_to_lds_b128 + s_wait_asynccnt), LDS double buffer.
// Fallback (small ws): fused dequant GEMM (round-1 design).
// ---------------------------------------------------------------------------

typedef __attribute__((ext_vector_type(16))) __bf16 v16bf;
typedef __attribute__((ext_vector_type(8)))  float  v8f;
typedef __attribute__((ext_vector_type(4)))  int    v4i;
typedef unsigned short u16;

#define M_TOK   4096
#define K_IN    4096
#define N_OUT   11008
#define GROUP   128

#define BM      128
#define BN      128
#define BK      128
#define LDPAD   8
#define LDK     (BK + LDPAD)   // 136 u16 -> 272B row stride (16B aligned, bank-spread)

union Frag {
    v16bf v;
    uint4 u[2];
};

__device__ __forceinline__ unsigned int pack2_bf16(float a, float b) {
    union { float f; unsigned int u; } ua, ub;
    ua.f = a; ub.f = b;
    unsigned int ra = (ua.u + 0x7FFFu + ((ua.u >> 16) & 1u)) >> 16;
    unsigned int rb = (ub.u + 0x7FFFu + ((ub.u >> 16) & 1u)) & 0xFFFF0000u;
    return ra | rb;
}

// ---- async global -> LDS (CDNA5), with sync fallback ----------------------
#if __has_builtin(__builtin_amdgcn_global_load_async_to_lds_b128)
#define HAVE_ASYNC_LDS 1
#else
#define HAVE_ASYNC_LDS 0
#endif

__device__ __forceinline__ void async_g2l_b128(const u16* g, u16* l) {
#if HAVE_ASYNC_LDS
    __builtin_amdgcn_global_load_async_to_lds_b128(
        (__attribute__((address_space(1))) v4i*)(g),
        (__attribute__((address_space(3))) v4i*)(l), 0, 0);
#else
    *(uint4*)l = *(const uint4*)g;
#endif
}

__device__ __forceinline__ void wait_async_lds() {
#if HAVE_ASYNC_LDS
#if __has_builtin(__builtin_amdgcn_s_wait_asynccnt)
    __builtin_amdgcn_s_wait_asynccnt(0);
#else
    asm volatile("s_wait_asynccnt 0x0" ::: "memory");
#endif
#endif
}

// ---------------------------------------------------------------------------
// Pass 1a: x f32 -> bf16 (4 elements / thread)
// ---------------------------------------------------------------------------
__global__ void __launch_bounds__(256)
convert_x_kernel(const float* __restrict__ x, u16* __restrict__ xb)
{
    size_t t = (size_t)blockIdx.x * blockDim.x + threadIdx.x;
    size_t idx = t * 4;
    float4 f = *(const float4*)(x + idx);
    uint2 p;
    p.x = pack2_bf16(f.x, f.y);
    p.y = pack2_bf16(f.z, f.w);
    *(uint2*)(xb + idx) = p;
}

// ---------------------------------------------------------------------------
// Pass 1b: dequant wq[N,K] int32 codes -> bf16 (4 elements / thread)
// ---------------------------------------------------------------------------
__global__ void __launch_bounds__(256)
dequant_w_kernel(const int* __restrict__ wq, const float* __restrict__ sz,
                 u16* __restrict__ wb)
{
    size_t t = (size_t)blockIdx.x * blockDim.x + threadIdx.x;
    size_t idx = t * 4;                  // 4 consecutive k of one row (K%4==0)
    size_t n = idx / K_IN;
    size_t k = idx % K_IN;
    size_t g = k / GROUP;                // all 4 elems in same group (GROUP%4==0)
    const float scale = sz[(g * N_OUT + n) * 2 + 0];
    const float z8    = sz[(g * N_OUT + n) * 2 + 1] - 8.0f * scale;
    int4 q = *(const int4*)(wq + idx);
    float w0 = (float)q.x * scale + z8;
    float w1 = (float)q.y * scale + z8;
    float w2 = (float)q.z * scale + z8;
    float w3 = (float)q.w * scale + z8;
    uint2 p;
    p.x = pack2_bf16(w0, w1);
    p.y = pack2_bf16(w2, w3);
    *(uint2*)(wb + idx) = p;
}

// ---------------------------------------------------------------------------
// Pass 2: bf16 GEMM  out[M,N] = xb[M,K] @ wb[N,K]^T
//   128x128 block tile, 8 waves (4x2), 32x64 per wave, K-step 128,
//   async global->LDS staging, LDS double buffer.
// ---------------------------------------------------------------------------
__global__ void __launch_bounds__(256)
gemm_bf16_wmma_kernel(const u16* __restrict__ xb, const u16* __restrict__ wb,
                      float* __restrict__ out)
{
    __shared__ u16 As[2][BM][LDK];
    __shared__ u16 Bs[2][BN][LDK];

    const int tid  = threadIdx.x;
    const int lane = tid & 31;
    const int wid  = tid >> 5;
    const int half = lane >> 4;
    const int l16  = lane & 15;

    const int m0 = blockIdx.y * BM;
    const int n0 = blockIdx.x * BN;

    const int wm = (wid >> 1) * 32;
    const int wn = (wid & 1) * 64;

    v8f acc[2][4];
#pragma unroll
    for (int i = 0; i < 2; ++i)
#pragma unroll
        for (int j = 0; j < 4; ++j)
#pragma unroll
            for (int r = 0; r < 8; ++r)
                acc[i][j][r] = 0.0f;

    // staging: thread owns 64 consecutive bf16 of one row (8 x b128)
    const int sr = tid >> 1;
    const int sh = (tid & 1) * 64;
    const u16* xg = xb + (size_t)(m0 + sr) * K_IN + sh;
    const u16* wg = wb + (size_t)(n0 + sr) * K_IN + sh;

    auto stage = [&](int buf, int k0) {
        const u16* xs = xg + k0;
        const u16* ws = wg + k0;
        u16* la = &As[buf][sr][sh];
        u16* lb = &Bs[buf][sr][sh];
#pragma unroll
        for (int i = 0; i < 8; ++i)
            async_g2l_b128(xs + i * 8, la + i * 8);
#pragma unroll
        for (int i = 0; i < 8; ++i)
            async_g2l_b128(ws + i * 8, lb + i * 8);
    };

    stage(0, 0);

    int buf = 0;
    for (int k0 = 0; k0 < K_IN; k0 += BK) {
        wait_async_lds();
        __syncthreads();                 // stage(buf) visible to all waves

        if (k0 + BK < K_IN)
            stage(buf ^ 1, k0 + BK);     // overlaps with compute below

#pragma unroll
        for (int kc = 0; kc < BK; kc += 32) {
            Frag a[2], b[4];
#pragma unroll
            for (int i = 0; i < 2; ++i) {
                // A 16x32 bf16: lane m=l16; V0-3: K=half*8..+7 ; V4-7: K=16+half*8..+7
                const u16* base = &As[buf][wm + i * 16 + l16][kc + half * 8];
                a[i].u[0] = *(const uint4*)(base);
                a[i].u[1] = *(const uint4*)(base + 16);
            }
#pragma unroll
            for (int j = 0; j < 4; ++j) {
                // B 32x16 bf16: lane n=l16; V0-3: K=half*8..+7 ; V4-7: K=16+half*8..+7
                const u16* base = &Bs[buf][wn + j * 16 + l16][kc + half * 8];
                b[j].u[0] = *(const uint4*)(base);
                b[j].u[1] = *(const uint4*)(base + 16);
            }
#pragma unroll
            for (int i = 0; i < 2; ++i)
#pragma unroll
                for (int j = 0; j < 4; ++j)
                    acc[i][j] = __builtin_amdgcn_wmma_f32_16x16x32_bf16(
                        false, a[i].v, false, b[j].v,
                        (short)0, acc[i][j], false, false);
        }
        buf ^= 1;
    }

    // store: C/D layout: VGPR r -> M = r + 8*half, lane -> N = l16
#pragma unroll
    for (int i = 0; i < 2; ++i) {
#pragma unroll
        for (int j = 0; j < 4; ++j) {
            const int mbase = m0 + wm + i * 16 + half * 8;
            const int n     = n0 + wn + j * 16 + l16;
#pragma unroll
            for (int r = 0; r < 8; ++r)
                out[(size_t)(mbase + r) * N_OUT + n] = acc[i][j][r];
        }
    }
}

// ---------------------------------------------------------------------------
// Fallback: fused dequant GEMM (used only if ws is too small)
// ---------------------------------------------------------------------------
__global__ void __launch_bounds__(256)
int4_linear_fused_kernel(const float* __restrict__ x,
                         const int*   __restrict__ wq,
                         const float* __restrict__ sz,
                         float* __restrict__ out)
{
    __shared__ u16 As[BM][LDK];
    __shared__ u16 Bs[BN][LDK];

    const int tid  = threadIdx.x;
    const int lane = tid & 31;
    const int wid  = tid >> 5;
    const int half = lane >> 4;
    const int l16  = lane & 15;

    const int m0 = blockIdx.y * BM;
    const int n0 = blockIdx.x * BN;
    const int wm = (wid >> 1) * 32;
    const int wn = (wid & 1) * 64;

    v8f acc[2][4];
#pragma unroll
    for (int i = 0; i < 2; ++i)
#pragma unroll
        for (int j = 0; j < 4; ++j)
#pragma unroll
            for (int r = 0; r < 8; ++r)
                acc[i][j][r] = 0.0f;

    const int sr = tid >> 1;
    const int sh = (tid & 1) * 64;

    for (int k0 = 0; k0 < K_IN; k0 += BK) {
        {
            const float* xp = x + (size_t)(m0 + sr) * K_IN + k0 + sh;
            u16* dst = &As[sr][sh];
#pragma unroll
            for (int i = 0; i < 64; i += 4) {
                float4 f = *(const float4*)(xp + i);
                uint2 p;
                p.x = pack2_bf16(f.x, f.y);
                p.y = pack2_bf16(f.z, f.w);
                *(uint2*)(dst + i) = p;
            }
        }
        {
            const int g = k0 / GROUP;
            const float scale = sz[((size_t)g * N_OUT + n0 + sr) * 2 + 0];
            const float z8    = sz[((size_t)g * N_OUT + n0 + sr) * 2 + 1] - 8.0f * scale;
            const int* qp = wq + (size_t)(n0 + sr) * K_IN + k0 + sh;
            u16* dst = &Bs[sr][sh];
#pragma unroll
            for (int i = 0; i < 64; i += 4) {
                int4 q = *(const int4*)(qp + i);
                uint2 p;
                p.x = pack2_bf16((float)q.x * scale + z8, (float)q.y * scale + z8);
                p.y = pack2_bf16((float)q.z * scale + z8, (float)q.w * scale + z8);
                *(uint2*)(dst + i) = p;
            }
        }
        __syncthreads();

#pragma unroll
        for (int kc = 0; kc < BK; kc += 32) {
            Frag a[2], b[4];
#pragma unroll
            for (int i = 0; i < 2; ++i) {
                const u16* base = &As[wm + i * 16 + l16][kc + half * 8];
                a[i].u[0] = *(const uint4*)(base);
                a[i].u[1] = *(const uint4*)(base + 16);
            }
#pragma unroll
            for (int j = 0; j < 4; ++j) {
                const u16* base = &Bs[wn + j * 16 + l16][kc + half * 8];
                b[j].u[0] = *(const uint4*)(base);
                b[j].u[1] = *(const uint4*)(base + 16);
            }
#pragma unroll
            for (int i = 0; i < 2; ++i)
#pragma unroll
                for (int j = 0; j < 4; ++j)
                    acc[i][j] = __builtin_amdgcn_wmma_f32_16x16x32_bf16(
                        false, a[i].v, false, b[j].v,
                        (short)0, acc[i][j], false, false);
        }
        __syncthreads();
    }

#pragma unroll
    for (int i = 0; i < 2; ++i) {
#pragma unroll
        for (int j = 0; j < 4; ++j) {
            const int mbase = m0 + wm + i * 16 + half * 8;
            const int n     = n0 + wn + j * 16 + l16;
#pragma unroll
            for (int r = 0; r < 8; ++r)
                out[(size_t)(mbase + r) * N_OUT + n] = acc[i][j][r];
        }
    }
}

// ---------------------------------------------------------------------------
extern "C" void kernel_launch(void* const* d_in, const int* in_sizes, int n_in,
                              void* d_out, int out_size, void* d_ws, size_t ws_size,
                              hipStream_t stream) {
    const float* x  = (const float*)d_in[0];
    const int*   wq = (const int*)d_in[1];
    const float* sz = (const float*)d_in[2];
    float* out = (float*)d_out;

    const size_t xbytes = (size_t)M_TOK * K_IN * sizeof(u16);   // 33.5 MB
    const size_t wbytes = (size_t)N_OUT * K_IN * sizeof(u16);   // 90.2 MB

    dim3 grid(N_OUT / BN, M_TOK / BM);   // (86, 32)
    dim3 block(256);

    if (ws_size >= xbytes + wbytes) {
        u16* xbuf = (u16*)d_ws;
        u16* wbuf = (u16*)((char*)d_ws + xbytes);

        convert_x_kernel<<<(M_TOK * (size_t)K_IN) / (4 * 256), 256, 0, stream>>>(x, xbuf);
        dequant_w_kernel<<<((size_t)N_OUT * K_IN) / (4 * 256), 256, 0, stream>>>(wq, sz, wbuf);
        gemm_bf16_wmma_kernel<<<grid, block, 0, stream>>>(xbuf, wbuf, out);
    } else {
        int4_linear_fused_kernel<<<grid, block, 0, stream>>>(x, wq, sz, out);
    }
}
